// SegField_57492432224427
// MI455X (gfx1250) — matrix operand order repaired and checked by reading, not compiled, not run on hardware
//
#include <hip/hip_runtime.h>
#include <hip/hip_bf16.h>

typedef __attribute__((ext_vector_type(16))) _Float16 v16h;
typedef __attribute__((ext_vector_type(8)))  _Float16 v8h;
typedef __attribute__((ext_vector_type(8)))  float    v8f;

namespace segfield {

constexpr int   kB     = 2;
constexpr int   kH     = 256;
constexpr int   kW     = 256;
constexpr int   kN     = kB * kH * kW;      // 131072
constexpr int   kH0    = 64;
constexpr int   kW0    = 64;
constexpr int   kCEmb  = 32;
constexpr int   kCPe   = 128;
constexpr int   kInDim = 202;
constexpr int   kK1    = 224;               // kInDim padded to 32
constexpr int   kL0A   = 512;
constexpr int   kL0B   = 256;
constexpr int   kHead  = 129;
constexpr int   kHeadP = 144;               // padded to 16
constexpr int   kF     = 128;
constexpr int   kFineIn= 330;
constexpr int   kKF    = 352;               // kFineIn padded to 32
constexpr int   kSel   = 26214;             // int(kN * 0.2)
constexpr int   kSelP  = 26224;             // padded to 16
constexpr float kBnInv = 0.99999500003749971f;  // 1/sqrt(1+1e-5)
constexpr float kTwoPi = 6.283185307179586f;

// ---------------------------------------------------------------------------
// A fragment (16x32 f16, ISA 7.12.2): lanes 0-15 hold M=lane, K={0..7,16..23};
// lanes 16-31 hold M=lane-16, K={8..15,24..31}. Two aligned 16B vector loads.
__device__ inline v16h combine_a(const v8h lo, const v8h hv) {
  return __builtin_shufflevector(lo, hv, 0, 1, 2, 3, 4, 5, 6, 7,
                                 8, 9, 10, 11, 12, 13, 14, 15);
}

// ---------------------------------------------------------------------------
__global__ void zero_i32_k(int* __restrict__ p, int n) {
  int i = blockIdx.x * blockDim.x + threadIdx.x;
  if (i < n) p[i] = 0;
}

// ---------------------------------------------------------------------------
// f32 row-major (ks x ns) -> f16 WMMA-fragment-shuffled (kp x np, zero pad).
// dst element i: frag f = i/512, lane = (i%512)/16, e = i%16;
// K = 32*kc + 8*(lane>>4) + (e&7) + 16*(e>>3), N = 16*nt + (lane&15), f = kc*(np/16)+nt.
__global__ void convert_w_frag_k(const float* __restrict__ src,
                                 _Float16* __restrict__ dst,
                                 int ks, int ns, int kp, int np) {
  int i = blockIdx.x * blockDim.x + threadIdx.x;
  if (i >= kp * np) return;
  const int f      = i >> 9;
  const int within = i & 511;
  const int lane   = within >> 4;
  const int e      = within & 15;
  const int ntiles = np >> 4;
  const int kc = f / ntiles;
  const int nt = f - kc * ntiles;
  const int K = kc * 32 + ((lane >> 4) << 3) + (e & 7) + ((e >> 3) << 4);
  const int n = (nt << 4) + (lane & 15);
  dst[i] = (K < ks && n < ns) ? (_Float16)src[(size_t)K * ns + n] : (_Float16)0.f;
}

// ---------------------------------------------------------------------------
// Bilinear upsample (half-pixel centers) + positional encoding -> feat f16
__global__ __launch_bounds__(256)
void featgen_k(const float* __restrict__ emb, const float* __restrict__ pe,
               _Float16* __restrict__ feat) {
  int n = blockIdx.x * blockDim.x + threadIdx.x;
  if (n >= kN) return;
  const int x = n % kW;
  const int y = (n / kW) % kH;
  const int b = n / (kW * kH);

  const float sy = (y + 0.5f) * ((float)kH0 / kH) - 0.5f;
  const float sx = (x + 0.5f) * ((float)kW0 / kW) - 0.5f;
  const int iy = (int)floorf(sy), ix = (int)floorf(sx);
  const float fy = sy - (float)iy, fx = sx - (float)ix;
  const int y0 = min(max(iy, 0), kH0 - 1), y1 = min(max(iy + 1, 0), kH0 - 1);
  const int x0 = min(max(ix, 0), kW0 - 1), x1 = min(max(ix + 1, 0), kW0 - 1);
  const float w00 = (1.f - fy) * (1.f - fx), w01 = (1.f - fy) * fx;
  const float w10 = fy * (1.f - fx),         w11 = fy * fx;

  _Float16* out = feat + (size_t)n * kK1;
#pragma unroll 4
  for (int c = 0; c < kCEmb; ++c) {
    const float* p = emb + (size_t)(b * kCEmb + c) * kH0 * kW0;
    float v = w00 * p[y0 * kW0 + x0] + w01 * p[y0 * kW0 + x1] +
              w10 * p[y1 * kW0 + x0] + w11 * p[y1 * kW0 + x1];
    out[c] = (_Float16)v;
  }
#pragma unroll 4
  for (int c = 0; c < kCPe; ++c) {
    const float* p = pe + (size_t)(b * kCPe + c) * kH0 * kW0;
    float v = w00 * p[y0 * kW0 + x0] + w01 * p[y0 * kW0 + x1] +
              w10 * p[y1 * kW0 + x0] + w11 * p[y1 * kW0 + x1];
    out[kCEmb + c] = (_Float16)v;
  }
  const float yc = -1.f + 2.f * (float)y / (float)(kH - 1);
  const float xc = -1.f + 2.f * (float)x / (float)(kW - 1);
  const float ay = kTwoPi * yc, ax = kTwoPi * xc;
  const int base = kCEmb + kCPe;  // 160
#pragma unroll
  for (int j = 0; j < 10; ++j) {
    const float f = exp2f((float)j * (10.f / 9.f));
    out[base + j]      = (_Float16)sinf(ay * f);
    out[base + 10 + j] = (_Float16)sinf(ax * f);
    out[base + 20 + j] = (_Float16)cosf(ay * f);
    out[base + 30 + j] = (_Float16)cosf(ax * f);
  }
  out[base + 40] = (_Float16)yc;
  out[base + 41] = (_Float16)xc;
#pragma unroll
  for (int c = kInDim; c < kK1; ++c) out[c] = (_Float16)0.f;
}

// ---------------------------------------------------------------------------
// Generic WMMA GEMM: Out = relu(bn(A @ W + bias)); wave owns a 16x64 tile.
// A: M x K (f16, row-major, K % 32 == 0), Wsh: fragment-shuffled K x Nout
// (Nout % 64 == 0). bias/gain/beta: f32[Nout]. Both operand streams use
// incremented base pointers so loads carry constant immediate offsets.
__global__ __launch_bounds__(256)
void gemm_bn_relu_k(const _Float16* __restrict__ A, const _Float16* __restrict__ Wsh,
                    const float* __restrict__ bias, const float* __restrict__ gain,
                    const float* __restrict__ beta, _Float16* __restrict__ Out,
                    int M, int K, int Nout) {
  const int wave = (blockIdx.x * blockDim.x + threadIdx.x) >> 5;
  const int lane = threadIdx.x & 31;
  const int ngroups = Nout >> 6;
  const int mtiles  = M >> 4;
  if (wave >= mtiles * ngroups) return;
  const int mt   = wave / ngroups;
  const int ng   = wave - mt * ngroups;
  const int row0 = mt << 4;
  const int col0 = ng << 6;
  const int ntiles = Nout >> 4;

  const _Float16* ap = A + (size_t)(row0 + (lane & 15)) * K + ((lane >> 4) << 3);
  const _Float16* bp = Wsh + ((size_t)(ng << 2) << 9) + (lane << 4);
  const size_t bstep = (size_t)ntiles << 9;  // halves to next K-chunk
  const int nchunks = K >> 5;

  v8f acc[4] = {};
  for (int kc = 0; kc < nchunks; ++kc) {
    if (kc + 1 < nchunks)
      __builtin_prefetch((const void*)(ap + 32), 0, 1);
    const v16h a  = combine_a(*(const v8h*)ap, *(const v8h*)(ap + 16));
    const v16h b0 = *(const v16h*)(bp);
    const v16h b1 = *(const v16h*)(bp + 512);
    const v16h b2 = *(const v16h*)(bp + 1024);
    const v16h b3 = *(const v16h*)(bp + 1536);
    acc[0] = __builtin_amdgcn_wmma_f32_16x16x32_f16(false, a, false, b0, (short)0, acc[0], false, false);
    acc[1] = __builtin_amdgcn_wmma_f32_16x16x32_f16(false, a, false, b1, (short)0, acc[1], false, false);
    acc[2] = __builtin_amdgcn_wmma_f32_16x16x32_f16(false, a, false, b2, (short)0, acc[2], false, false);
    acc[3] = __builtin_amdgcn_wmma_f32_16x16x32_f16(false, a, false, b3, (short)0, acc[3], false, false);
    ap += 32;
    bp += bstep;
  }

  const int cn    = lane & 15;
  const int rbase = row0 + ((lane >> 4) << 3);  // lanes 16-31 hold M = 8..15
#pragma unroll
  for (int t = 0; t < 4; ++t) {
    const int col  = col0 + (t << 4) + cn;
    const float bv = bias[col];
    const float sc = gain[col] * kBnInv;
    const float sh = beta[col];
#pragma unroll
    for (int r = 0; r < 8; ++r) {
      float v = (acc[t][r] + bv) * sc + sh;
      v = v > 0.f ? v : 0.f;
      Out[(size_t)(rbase + r) * Nout + col] = (_Float16)v;
    }
  }
}

// ---------------------------------------------------------------------------
// Head: head = X2 @ w2 + b2 (129 cols padded to 144), fused dropout-pair
// epilogue: seg_coarse = 2*h*m2 (col 0), seg_feat = 2*h*m2 (cols 1..128),
// var = mean(2*h^2*(m1 xor m2)). One wave per 16-row strip; the 8 A
// fragments are hoisted into registers and reused across the 9 N-tiles.
__global__ __launch_bounds__(256)
void head_var_k(const _Float16* __restrict__ X2, const _Float16* __restrict__ W2sh,
                const float* __restrict__ b2, const int* __restrict__ dm1,
                const int* __restrict__ dm2, float* __restrict__ segc,
                _Float16* __restrict__ segf, float* __restrict__ var) {
  const int wave = (blockIdx.x * blockDim.x + threadIdx.x) >> 5;
  const int lane = threadIdx.x & 31;
  if (wave >= kN / 16) return;
  const int row0  = wave << 4;
  const int cn    = lane & 15;
  const int rbase = row0 + ((lane >> 4) << 3);
  constexpr int kNT = kHeadP >> 4;  // 9 N-tiles
  constexpr int kKC = kL0B / 32;    // 8 K-chunks

  const _Float16* ap = X2 + (size_t)(row0 + cn) * kL0B + ((lane >> 4) << 3);
  v16h afr[kKC];
#pragma unroll
  for (int kc = 0; kc < kKC; ++kc)
    afr[kc] = combine_a(*(const v8h*)(ap + kc * 32), *(const v8h*)(ap + kc * 32 + 16));

  float varpart[8] = {0.f, 0.f, 0.f, 0.f, 0.f, 0.f, 0.f, 0.f};

  for (int t = 0; t < kNT; ++t) {
    const _Float16* bp = W2sh + ((size_t)t << 9) + (lane << 4);
    v8f acc = {};
#pragma unroll
    for (int kc = 0; kc < kKC; ++kc) {
      const v16h b = *(const v16h*)(bp + (size_t)kc * (kNT << 9));
      acc = __builtin_amdgcn_wmma_f32_16x16x32_f16(false, afr[kc], false, b, (short)0, acc, false, false);
    }
    const int col = (t << 4) + cn;
    if (col < kHead) {
      const float bv = b2[col];
#pragma unroll
      for (int r = 0; r < 8; ++r) {
        const int row = rbase + r;
        const float h = acc[r] + bv;
        const int m1 = dm1[(size_t)row * kHead + col];
        const int m2 = dm2[(size_t)row * kHead + col];
        const float o2 = (m2 != 0) ? 2.f * h : 0.f;
        if ((m1 != 0) != (m2 != 0)) varpart[r] += 2.f * h * h;
        if (col == 0) segc[row] = o2;
        else          segf[(size_t)row * kF + (col - 1)] = (_Float16)o2;
      }
    }
  }
  // Row sums: within each 16-lane half, every lane of VGPR r holds row rbase+r.
#pragma unroll
  for (int r = 0; r < 8; ++r) {
    float v = varpart[r];
    v += __shfl_xor(v, 1, 16);
    v += __shfl_xor(v, 2, 16);
    v += __shfl_xor(v, 4, 16);
    v += __shfl_xor(v, 8, 16);
    if (cn == 0) var[rbase + r] = v * (1.f / (float)kHead);
  }
}

// ---------------------------------------------------------------------------
// Top-k selection: radix histogram over the float bit pattern (monotone for
// var >= 0), threshold scan, capped atomic compaction -> exactly kSel indices.
__global__ void hist_k(const float* __restrict__ var, int* __restrict__ hist) {
  int n = blockIdx.x * blockDim.x + threadIdx.x;
  if (n >= kN) return;
  unsigned b = __float_as_uint(var[n]) >> 20;
  if (b > 2047u) b = 2047u;
  atomicAdd(&hist[b], 1);
}

__global__ void select_threshold_k(const int* __restrict__ hist, int* __restrict__ thr) {
  if (blockIdx.x != 0 || threadIdx.x != 0) return;
  int acc = 0, T = 0, rem = kSel;
  for (int b = 2047; b >= 0; --b) {
    const int c = hist[b];
    if (acc + c >= kSel) { T = b; rem = kSel - acc; break; }
    acc += c;
    if (b == 0) { T = 0; rem = kSel - acc; }
  }
  thr[0] = T;
  thr[1] = rem;
}

__global__ void compact_k(const float* __restrict__ var, const int* __restrict__ thr,
                          int* __restrict__ sel, int* __restrict__ cnts) {
  int n = blockIdx.x * blockDim.x + threadIdx.x;
  if (n >= kN) return;
  unsigned b = __float_as_uint(var[n]) >> 20;
  if (b > 2047u) b = 2047u;
  const int T = thr[0], rem = thr[1];
  if ((int)b > T) {
    int p = atomicAdd(&cnts[0], 1);
    sel[p] = n;
  } else if ((int)b == T) {
    int r = atomicAdd(&cnts[1], 1);
    if (r < rem) {
      int p = atomicAdd(&cnts[0], 1);
      sel[p] = n;
    }
  }
}

// ---------------------------------------------------------------------------
// Gather fine_in rows: [feat(202) | seg_feat(128) | pad(22)], zero tail rows.
__global__ void gather_fine_k(const _Float16* __restrict__ feat,
                              const _Float16* __restrict__ segf,
                              const int* __restrict__ sel,
                              _Float16* __restrict__ fine) {
  const int i = blockIdx.x;
  const int c = threadIdx.x;  // 0..351
  _Float16 v = (_Float16)0.f;
  if (i < kSel) {
    const int s = sel[i];
    if (c < kInDim)        v = feat[(size_t)s * kK1 + c];
    else if (c < kFineIn)  v = segf[(size_t)s * kF + (c - kInDim)];
  }
  fine[(size_t)i * kKF + c] = v;
}

// ---------------------------------------------------------------------------
// Output: coarse image to both halves, then scatter fine z = y3 @ w3b + b3b.
__global__ void copy_out_k(const float* __restrict__ segc, float* __restrict__ out) {
  int n = blockIdx.x * blockDim.x + threadIdx.x;
  if (n >= kN) return;
  const float v = segc[n];
  out[n] = v;
  out[kN + n] = v;
}

__global__ void fine_out_k(const _Float16* __restrict__ y3, const float* __restrict__ w3b,
                           const float* __restrict__ b3b, const int* __restrict__ sel,
                           float* __restrict__ out) {
  int i = blockIdx.x * blockDim.x + threadIdx.x;
  if (i >= kSel) return;
  float s = b3b[0];
  const _Float16* p = y3 + (size_t)i * 64;
#pragma unroll
  for (int j = 0; j < 64; ++j) s += (float)p[j] * w3b[j];
  out[kN + sel[i]] = s;
}

}  // namespace segfield

// ---------------------------------------------------------------------------
extern "C" void kernel_launch(void* const* d_in, const int* in_sizes, int n_in,
                              void* d_out, int out_size, void* d_ws, size_t ws_size,
                              hipStream_t stream) {
  using namespace segfield;
  (void)in_sizes; (void)n_in; (void)out_size; (void)ws_size;

  const float* emb  = (const float*)d_in[0];
  const float* pe   = (const float*)d_in[1];
  const int*   dm1  = (const int*)d_in[2];
  const int*   dm2  = (const int*)d_in[3];
  const float* w0a  = (const float*)d_in[4];
  const float* b0a  = (const float*)d_in[5];
  const float* g0a  = (const float*)d_in[6];
  const float* be0a = (const float*)d_in[7];
  const float* w0b  = (const float*)d_in[8];
  const float* b0b  = (const float*)d_in[9];
  const float* g0b  = (const float*)d_in[10];
  const float* be0b = (const float*)d_in[11];
  const float* w2   = (const float*)d_in[12];
  const float* b2   = (const float*)d_in[13];
  const float* w1a  = (const float*)d_in[14];
  const float* b1a  = (const float*)d_in[15];
  const float* g1a  = (const float*)d_in[16];
  const float* be1a = (const float*)d_in[17];
  const float* w1b  = (const float*)d_in[18];
  const float* b1b  = (const float*)d_in[19];
  const float* g1b  = (const float*)d_in[20];
  const float* be1b = (const float*)d_in[21];
  const float* w3a  = (const float*)d_in[22];
  const float* b3a  = (const float*)d_in[23];
  const float* g3a  = (const float*)d_in[24];
  const float* be3a = (const float*)d_in[25];
  const float* w3b  = (const float*)d_in[26];
  const float* b3b  = (const float*)d_in[27];
  float* out = (float*)d_out;

  // Workspace carve-out (256B aligned).
  char* ws = (char*)d_ws;
  size_t off = 0;
  auto alloc = [&](size_t bytes) -> void* {
    void* p = ws + off;
    off = (off + bytes + 255) & ~(size_t)255;
    return p;
  };
  _Float16* feat = (_Float16*)alloc((size_t)kN * kK1 * 2);
  _Float16* X1   = (_Float16*)alloc((size_t)kN * kL0A * 2);
  _Float16* X2   = (_Float16*)alloc((size_t)kN * kL0B * 2);
  _Float16* segf = (_Float16*)alloc((size_t)kN * kF * 2);
  float*    segc = (float*)alloc((size_t)kN * 4);
  float*    var  = (float*)alloc((size_t)kN * 4);
  _Float16* w0aH = (_Float16*)alloc((size_t)kK1 * kL0A * 2);
  _Float16* w0bH = (_Float16*)alloc((size_t)kL0A * kL0B * 2);
  _Float16* w2H  = (_Float16*)alloc((size_t)kL0B * kHeadP * 2);
  _Float16* w1aH = (_Float16*)alloc((size_t)kKF * kF * 2);
  _Float16* w1bH = (_Float16*)alloc((size_t)kF * kF * 2);
  _Float16* w3aH = (_Float16*)alloc((size_t)kF * 64 * 2);
  int*      ctrl = (int*)alloc((size_t)(2048 + 4) * 4);  // hist | thr[2] | cnts[2]
  int*      sel  = (int*)alloc((size_t)kSelP * 4);
  // Fine-stage buffers overlaid on X1 (dead after GEMM2).
  char* fb = (char*)X1;
  _Float16* fineIn = (_Float16*)fb; fb += (size_t)kSelP * kKF * 2;
  _Float16* y1     = (_Float16*)fb; fb += (size_t)kSelP * kF * 2;
  _Float16* y2     = (_Float16*)fb; fb += (size_t)kSelP * kF * 2;
  _Float16* y3     = (_Float16*)fb;

  int* thr  = ctrl + 2048;
  int* cnts = ctrl + 2050;

  auto cdiv = [](int a, int b) { return (a + b - 1) / b; };

  // 0) clear histogram + counters
  zero_i32_k<<<cdiv(2052, 256), 256, 0, stream>>>(ctrl, 2052);

  // 1) weight conversion to f16, fragment-shuffled for single-load B operands
  auto conv = [&](const float* s, _Float16* d, int ks, int ns, int kp, int np) {
    convert_w_frag_k<<<cdiv(kp * np, 256), 256, 0, stream>>>(s, d, ks, ns, kp, np);
  };
  conv(w0a, w0aH, kInDim, kL0A, kK1, kL0A);
  conv(w0b, w0bH, kL0A, kL0B, kL0A, kL0B);
  conv(w2,  w2H,  kL0B, kHead, kL0B, kHeadP);
  conv(w1a, w1aH, kFineIn, kF, kKF, kF);
  conv(w1b, w1bH, kF, kF, kF, kF);
  conv(w3a, w3aH, kF, 64, kF, 64);

  // 2) feature generation (bilinear upsample + posenc)
  featgen_k<<<kN / 256, 256, 0, stream>>>(emb, pe, feat);

  // 3) trunk MLP via WMMA
  {
    int waves = (kN / 16) * (kL0A / 64);
    gemm_bn_relu_k<<<cdiv(waves * 32, 256), 256, 0, stream>>>(
        feat, w0aH, b0a, g0a, be0a, X1, kN, kK1, kL0A);
  }
  {
    int waves = (kN / 16) * (kL0B / 64);
    gemm_bn_relu_k<<<cdiv(waves * 32, 256), 256, 0, stream>>>(
        X1, w0bH, b0b, g0b, be0b, X2, kN, kL0A, kL0B);
  }

  // 4) head + dropout-pair variance
  {
    int waves = kN / 16;
    head_var_k<<<cdiv(waves * 32, 256), 256, 0, stream>>>(
        X2, w2H, b2, dm1, dm2, segc, segf, var);
  }

  // 5) top-k selection
  hist_k<<<cdiv(kN, 256), 256, 0, stream>>>(var, ctrl);
  select_threshold_k<<<1, 32, 0, stream>>>(ctrl, thr);
  compact_k<<<cdiv(kN, 256), 256, 0, stream>>>(var, thr, sel, cnts);

  // 6) fine MLP (rows padded to kSelP)
  gather_fine_k<<<kSelP, kKF, 0, stream>>>(feat, segf, sel, fineIn);
  {
    int waves = (kSelP / 16) * (kF / 64);
    gemm_bn_relu_k<<<cdiv(waves * 32, 256), 256, 0, stream>>>(
        fineIn, w1aH, b1a, g1a, be1a, y1, kSelP, kKF, kF);
    gemm_bn_relu_k<<<cdiv(waves * 32, 256), 256, 0, stream>>>(
        y1, w1bH, b1b, g1b, be1b, y2, kSelP, kF, kF);
  }
  {
    int waves = (kSelP / 16) * (64 / 64);
    gemm_bn_relu_k<<<cdiv(waves * 32, 256), 256, 0, stream>>>(
        y2, w3aH, b3a, g3a, be3a, y3, kSelP, kF, 64);
  }

  // 7) outputs
  copy_out_k<<<cdiv(kN, 256), 256, 0, stream>>>(segc, out);
  fine_out_k<<<cdiv(kSel, 256), 256, 0, stream>>>(y3, w3b, b3b, sel, out);
}
